// WSDModel_2903397892247
// MI455X (gfx1250) — matrix operand order repaired and checked by reading, not compiled, not run on hardware
//
#include <hip/hip_runtime.h>

// ---------------- problem constants ----------------
#define B_ 8
#define N_ 1024
#define D_ 512
#define Y_ 20000
#define LN_EPS 1e-5f

typedef __bf16 bf16;
typedef __attribute__((ext_vector_type(16))) __bf16 v16bf;
typedef __attribute__((ext_vector_type(8)))  float  v8f;
typedef __attribute__((ext_vector_type(4)))  unsigned int u32x4;

union FragBF {
    v16bf v;
    u32x4 q[2];
};

// Block tile: 128 (M) x 128 (N), K-step 32. 256 threads = 8 wave32.
// Wave w computes rows [128*by + 16w, +16) x cols [128*bx, +128) as 8 WMMA tiles.
#define TM 128
#define TN 128
#define TK 32

// ---- gfx1250 async memory->LDS path (guarded; sync staging as fallback) ----
#if __has_builtin(__builtin_amdgcn_global_load_async_to_lds_b128) && \
    __has_builtin(__builtin_amdgcn_s_wait_asynccnt)
#define ASYNC_LDS 1
typedef int v4i_t __attribute__((vector_size(16)));
typedef __attribute__((address_space(1))) v4i_t* gas_ptr;   // global int4*
typedef __attribute__((address_space(3))) v4i_t* las_ptr;   // LDS int4*
#else
#define ASYNC_LDS 0
#endif

__device__ __forceinline__ void cp16(const bf16* __restrict__ g, bf16* l)
{
#if ASYNC_LDS
    __builtin_amdgcn_global_load_async_to_lds_b128((gas_ptr)g, (las_ptr)l, 0, 0);
#else
    *(u32x4*)l = *(const u32x4*)g;
#endif
}

__device__ __forceinline__ void async_publish_wait()
{
#if ASYNC_LDS
    __builtin_amdgcn_s_wait_asynccnt(0);
#endif
}

// ---------------------------------------------------------------------------
// Stage one K-step tile pair into LDS (A: 128x32, B: 128x32 bf16).
// Out-of-range B rows are clamped (their garbage columns are never stored).
// ---------------------------------------------------------------------------
__device__ __forceinline__ void stage_tiles(
    const bf16* __restrict__ A, const bf16* __restrict__ Bm,
    int lda, int ldb, int mBlk, int nBlk, int NcM1,
    int k0, bf16* As, bf16* Bs, int t)
{
    #pragma unroll
    for (int i = t; i < TM * 4; i += 256) {          // 512 x 16B chunks
        int row = i >> 2, c = i & 3;
        cp16(A + (size_t)(mBlk + row) * lda + k0 + c * 8, As + row * TK + c * 8);
    }
    #pragma unroll
    for (int i = t; i < TN * 4; i += 256) {          // 512 x 16B chunks
        int row = i >> 2, c = i & 3;
        int grow = nBlk + row;
        grow = grow > NcM1 ? NcM1 : grow;
        cp16(Bm + (size_t)grow * ldb + k0 + c * 8, Bs + row * TK + c * 8);
    }
}

// ---------------------------------------------------------------------------
// Core: double-buffered block GEMM  C[128x128] += A[128xK] * B[128xK]^T
// (bf16 operands, f32 accum). acc[8] per wave in WMMA C/D layout.
// All 9 fragments are loaded up front so the 8 WMMAs issue back-to-back
// behind a single DScnt wait.
// ---------------------------------------------------------------------------
__device__ __forceinline__ void gemm_block_nt(
    const bf16* __restrict__ A, const bf16* __restrict__ Bm,
    int K, int lda, int ldb, int mBlk, int nBlk, int Nc,
    bf16* AsBuf, bf16* BsBuf, v8f acc[8])
{
    const int t    = threadIdx.x;
    const int w    = t >> 5;
    const int lane = t & 31;
    const int r    = lane & 15;   // row/col within 16
    const int hi   = lane >> 4;   // half-wave select
    const int NcM1 = Nc - 1;

    stage_tiles(A, Bm, lda, ldb, mBlk, nBlk, NcM1, 0, AsBuf, BsBuf, t);
    async_publish_wait();
    __syncthreads();

    const int nSteps = K / TK;
    for (int s = 0; s < nSteps; ++s) {
        const int p = s & 1;
        const bf16* Ac = AsBuf + p * (TM * TK);
        const bf16* Bc = BsBuf + p * (TN * TK);

        if (s + 1 < nSteps) {
            stage_tiles(A, Bm, lda, ldb, mBlk, nBlk, NcM1,
                        (s + 1) * TK,
                        AsBuf + (p ^ 1) * (TM * TK),
                        BsBuf + (p ^ 1) * (TN * TK), t);
        }
#if !ASYNC_LDS
        if (s + 1 < nSteps)
            __builtin_prefetch(A + (size_t)(mBlk + (t & 127)) * lda + (s + 1) * TK, 0, 0);
#endif

        // ---- A fragment: lane<16 -> K 0..15 contiguous; lane>=16 -> K 16..31 ----
        FragBF a;
        const bf16* arow = Ac + (w * 16 + r) * TK;
        a.q[0] = *(const u32x4*)(arow + hi * 16);
        a.q[1] = *(const u32x4*)(arow + hi * 16 + 8);

        // ---- all 8 B fragments first (distinct regs -> one dscnt wait) ----
        FragBF bfr[8];
        #pragma unroll
        for (int tn = 0; tn < 8; ++tn) {
            // lane<16 -> col r with K 0..15 contiguous; lane>=16 -> K 16..31
            const bf16* brow = Bc + (tn * 16 + r) * TK + hi * 16;
            bfr[tn].q[0] = *(const u32x4*)(brow);
            bfr[tn].q[1] = *(const u32x4*)(brow + 8);
        }

        // ---- 8 back-to-back WMMAs (independent accumulators) ----
        #pragma unroll
        for (int tn = 0; tn < 8; ++tn) {
            acc[tn] = __builtin_amdgcn_wmma_f32_16x16x32_bf16(
                false, a.v, false, bfr[tn].v, (short)0, acc[tn], false, false);
        }

        async_publish_wait();   // next-tile async copies done before publishing
        __syncthreads();
    }
}

// ---------------------------------------------------------------------------
// Generic batched NT GEMM, optional f32 (RT or NT stores) and/or bf16 stores.
// ---------------------------------------------------------------------------
__global__ __launch_bounds__(256)
void gemm_nt_kernel(const bf16* __restrict__ A, const bf16* __restrict__ Bm,
                    float* Cf, bf16* Cb, int ntStore,
                    int M, int Nc, int K, int lda, int ldb, int ldc,
                    long long sA, long long sB, long long sC)
{
    __shared__ bf16 As[2 * TM * TK];   // 16 KB
    __shared__ bf16 Bs[2 * TN * TK];   // 16 KB

    const int nBlk = blockIdx.x * TN;
    const int mBlk = blockIdx.y * TM;
    const long long z = blockIdx.z;
    const bf16* Ap = A + z * sA;
    const bf16* Bp = Bm + z * sB;

    v8f acc[8];
    #pragma unroll
    for (int i = 0; i < 8; ++i)
        acc[i] = (v8f){0.f,0.f,0.f,0.f,0.f,0.f,0.f,0.f};

    gemm_block_nt(Ap, Bp, K, lda, ldb, mBlk, nBlk, Nc, As, Bs, acc);

    const int lane = threadIdx.x & 31;
    const int w    = threadIdx.x >> 5;
    const int r    = lane & 15;
    const int hi   = lane >> 4;
    const int rBase = mBlk + w * 16 + hi * 8;

    #pragma unroll
    for (int tn = 0; tn < 8; ++tn) {
        int col = nBlk + tn * 16 + r;
        if (col < Nc) {
            #pragma unroll
            for (int v = 0; v < 8; ++v) {
                size_t idx = (size_t)(rBase + v) * (size_t)ldc + col + (size_t)z * sC;
                float x = acc[tn][v];
                if (Cf) {
                    if (ntStore) __builtin_nontemporal_store(x, &Cf[idx]);
                    else         Cf[idx] = x;
                }
                if (Cb) Cb[idx] = (bf16)x;
            }
        }
    }
}

// ---------------------------------------------------------------------------
// scores[b,i,j] = Q[b,i,:] . X[b,j,:]  - (i-j)^2 ;  masked keys -> -1e30
// Written directly into the A-output region (softmaxed in place later).
// ---------------------------------------------------------------------------
__global__ __launch_bounds__(256)
void scores_kernel(const bf16* __restrict__ Qb, const bf16* __restrict__ Xb,
                   const int* __restrict__ Ms, float* __restrict__ S)
{
    __shared__ bf16 As[2 * TM * TK];
    __shared__ bf16 Bs[2 * TN * TK];

    const int nBlk = blockIdx.x * TN;   // j tile
    const int mBlk = blockIdx.y * TM;   // i tile
    const int b    = blockIdx.z;

    const bf16* Ap = Qb + (size_t)b * N_ * D_;
    const bf16* Bp = Xb + (size_t)b * N_ * D_;
    const int*  Mb = Ms + (size_t)b * N_;
    float* Sb = S + (size_t)b * N_ * N_;

    v8f acc[8];
    #pragma unroll
    for (int i = 0; i < 8; ++i)
        acc[i] = (v8f){0.f,0.f,0.f,0.f,0.f,0.f,0.f,0.f};

    gemm_block_nt(Ap, Bp, D_, D_, D_, mBlk, nBlk, N_, As, Bs, acc);

    const int lane = threadIdx.x & 31;
    const int w    = threadIdx.x >> 5;
    const int r    = lane & 15;
    const int hi   = lane >> 4;
    const int rBase = mBlk + w * 16 + hi * 8;

    #pragma unroll
    for (int tn = 0; tn < 8; ++tn) {
        int j = nBlk + tn * 16 + r;
        bool keyok = (Mb[j] != 0);
        #pragma unroll
        for (int v = 0; v < 8; ++v) {
            int i = rBase + v;
            float d = (float)(i - j);
            float val = acc[tn][v] - d * d;
            if (!keyok) val = -1e30f;
            Sb[(size_t)i * N_ + j] = val;
        }
    }
}

// ---------------------------------------------------------------------------
// Row softmax over N_ (in place, f32) + bf16 copy for the next GEMM.
// ---------------------------------------------------------------------------
__global__ __launch_bounds__(256)
void softmax_kernel(float* __restrict__ S, bf16* __restrict__ Ab)
{
    const size_t row = blockIdx.x;              // b*N + i
    float* p  = S  + row * N_;
    bf16*  pb = Ab + row * N_;
    const int t = threadIdx.x;

    float v[4];
    float m = -3.4e38f;
    #pragma unroll
    for (int k = 0; k < 4; ++k) { v[k] = p[t + 256 * k]; m = fmaxf(m, v[k]); }

    __shared__ float red[256];
    red[t] = m; __syncthreads();
    for (int s = 128; s; s >>= 1) { if (t < s) red[t] = fmaxf(red[t], red[t + s]); __syncthreads(); }
    m = red[0]; __syncthreads();

    float sum = 0.f;
    #pragma unroll
    for (int k = 0; k < 4; ++k) { v[k] = __expf(v[k] - m); sum += v[k]; }
    red[t] = sum; __syncthreads();
    for (int s = 128; s; s >>= 1) { if (t < s) red[t] += red[t + s]; __syncthreads(); }
    const float inv = 1.0f / red[0];

    #pragma unroll
    for (int k = 0; k < 4; ++k) {
        float a = v[k] * inv;
        p[t + 256 * k]  = a;
        pb[t + 256 * k] = (bf16)a;
    }
}

// ---------------------------------------------------------------------------
// H = Qc + X; LayerNorm over D_; bf16 result for logits GEMM.
// ---------------------------------------------------------------------------
__global__ __launch_bounds__(256)
void add_ln_kernel(const float* __restrict__ Qc, const float* __restrict__ Xf,
                   const float* __restrict__ g, const float* __restrict__ bta,
                   bf16* __restrict__ Hn)
{
    const size_t row = blockIdx.x;
    const float* q = Qc + row * D_;
    const float* x = Xf + row * D_;
    const int t = threadIdx.x;

    float h0 = q[t] + x[t];
    float h1 = q[t + 256] + x[t + 256];

    __shared__ float rs[256], rq[256];
    rs[t] = h0 + h1;
    rq[t] = h0 * h0 + h1 * h1;
    __syncthreads();
    for (int s = 128; s; s >>= 1) {
        if (t < s) { rs[t] += rs[t + s]; rq[t] += rq[t + s]; }
        __syncthreads();
    }
    const float mu  = rs[0] * (1.0f / D_);
    const float var = rq[0] * (1.0f / D_) - mu * mu;
    const float inv = rsqrtf(var + LN_EPS);

    Hn[row * D_ + t]       = (bf16)((h0 - mu) * inv * g[t]       + bta[t]);
    Hn[row * D_ + t + 256] = (bf16)((h1 - mu) * inv * g[t + 256] + bta[t + 256]);
}

// ---------------------------------------------------------------------------
// Embedding gather: X f32, X bf16 (row-major [B*N, D]) and X^T bf16 [b][d][n].
// ---------------------------------------------------------------------------
__global__ __launch_bounds__(256)
void gather_kernel(const int* __restrict__ Ms, const float* __restrict__ Ev,
                   float* __restrict__ Xf, bf16* __restrict__ Xb, bf16* __restrict__ Xt)
{
    const size_t row = blockIdx.x;          // b*N + n
    const int b = (int)(row >> 10);
    const int n = (int)(row & (N_ - 1));
    const int tok = Ms[row];
    const float* src = Ev + (size_t)tok * D_;
    const int t = threadIdx.x;
    #pragma unroll
    for (int k = 0; k < 2; ++k) {
        int d = t + 256 * k;
        float v = src[d];
        Xf[row * D_ + d] = v;
        bf16 bv = (bf16)v;
        Xb[row * D_ + d] = bv;
        Xt[(size_t)b * D_ * N_ + (size_t)d * N_ + n] = bv;
    }
}

// ---------------------------------------------------------------------------
// Weight prep: W_A, W_O transposed to [e,d] bf16; E_y converted to bf16.
// ---------------------------------------------------------------------------
__global__ __launch_bounds__(256)
void prep_kernel(const float* __restrict__ WA, const float* __restrict__ WO,
                 const float* __restrict__ Ey,
                 bf16* __restrict__ WAt, bf16* __restrict__ WOt, bf16* __restrict__ Eyb)
{
    const size_t idx    = (size_t)blockIdx.x * 256 + threadIdx.x;
    const size_t stride = (size_t)gridDim.x * 256;
    for (size_t i = idx; i < (size_t)D_ * D_; i += stride) {
        size_t d = i >> 9, e = i & (D_ - 1);
        WAt[e * D_ + d] = (bf16)WA[i];
        WOt[e * D_ + d] = (bf16)WO[i];
    }
    for (size_t i = idx; i < (size_t)Y_ * D_; i += stride)
        Eyb[i] = (bf16)Ey[i];
}

// ---------------------------------------------------------------------------
extern "C" void kernel_launch(void* const* d_in, const int* in_sizes, int n_in,
                              void* d_out, int out_size, void* d_ws, size_t ws_size,
                              hipStream_t stream)
{
    const int*   Ms    = (const int*)d_in[0];
    const float* Ev    = (const float*)d_in[1];
    const float* Ey    = (const float*)d_in[2];
    const float* WA    = (const float*)d_in[3];
    const float* WO    = (const float*)d_in[4];
    const float* gamma = (const float*)d_in[5];
    const float* beta  = (const float*)d_in[6];

    float* out    = (float*)d_out;
    float* logits = out;                                   // [B,N,Y]
    float* Aout   = out + (size_t)B_ * N_ * Y_;            // [B,N,N]

    const size_t R = (size_t)B_ * N_;                      // 8192 rows
    char* ws = (char*)d_ws;
    size_t off = 0;
    float* Xf  = (float*)(ws + off); off += R * D_ * sizeof(float);
    bf16*  Xb  = (bf16*) (ws + off); off += R * D_ * sizeof(bf16);
    bf16*  Xt  = (bf16*) (ws + off); off += (size_t)B_ * D_ * N_ * sizeof(bf16);
    bf16*  WAt = (bf16*) (ws + off); off += (size_t)D_ * D_ * sizeof(bf16);
    bf16*  WOt = (bf16*) (ws + off); off += (size_t)D_ * D_ * sizeof(bf16);
    bf16*  Eyb = (bf16*) (ws + off); off += (size_t)Y_ * D_ * sizeof(bf16);
    bf16*  Qb  = (bf16*) (ws + off); off += R * D_ * sizeof(bf16);
    bf16*  Ab  = (bf16*) (ws + off); off += R * N_ * sizeof(bf16);
    bf16*  AXb = (bf16*) (ws + off); off += R * D_ * sizeof(bf16);
    float* Qcf = (float*)(ws + off); off += R * D_ * sizeof(float);
    bf16*  Hnb = (bf16*) (ws + off); off += R * D_ * sizeof(bf16);
    (void)ws_size; (void)in_sizes; (void)n_in; (void)out_size;

    // 1. weight prep (bf16 transposes / conversions)
    prep_kernel<<<4096, 256, 0, stream>>>(WA, WO, Ey, WAt, WOt, Eyb);

    // 2. embedding gather
    gather_kernel<<<(unsigned)R, 256, 0, stream>>>(Ms, Ev, Xf, Xb, Xt);

    // 3. Q = X @ W_A           [8192,512] = [8192,512] x [512,512]^T
    gemm_nt_kernel<<<dim3(D_ / TN, (unsigned)(R / TM), 1), 256, 0, stream>>>(
        Xb, WAt, nullptr, Qb, 0, (int)R, D_, D_, D_, D_, D_, 0, 0, 0);

    // 4. scores + bias + mask  -> A-region of d_out
    scores_kernel<<<dim3(N_ / TN, N_ / TM, B_), 256, 0, stream>>>(Qb, Xb, Ms, Aout);

    // 5. softmax (in place) + bf16 copy
    softmax_kernel<<<(unsigned)R, 256, 0, stream>>>(Aout, Ab);

    // 6. AX = A @ X            per batch: [1024,512] = [1024,1024] x [512,1024]^T
    gemm_nt_kernel<<<dim3(D_ / TN, N_ / TM, B_), 256, 0, stream>>>(
        Ab, Xt, nullptr, AXb, 0, N_, D_, N_, N_, N_, D_,
        (long long)N_ * N_, (long long)D_ * N_, (long long)N_ * D_);

    // 7. Qc = AX @ W_O         [8192,512]
    gemm_nt_kernel<<<dim3(D_ / TN, (unsigned)(R / TM), 1), 256, 0, stream>>>(
        AXb, WOt, Qcf, nullptr, 0, (int)R, D_, D_, D_, D_, D_, 0, 0, 0);

    // 8. residual + LayerNorm -> Hn (bf16)
    add_ln_kernel<<<(unsigned)R, 256, 0, stream>>>(Qcf, Xf, gamma, beta, Hnb);

    // 9. logits = Hn @ E_y^T   [8192,20000]  (non-temporal stores: 655 MB stream)
    gemm_nt_kernel<<<dim3((Y_ + TN - 1) / TN, (unsigned)(R / TM), 1), 256, 0, stream>>>(
        Hnb, Eyb, logits, nullptr, 1, (int)R, Y_, D_, D_, D_, Y_, 0, 0, 0);
}
